// RelativeAxialAttention_75496935129304
// MI455X (gfx1250) — matrix operand order repaired and checked by reading, not compiled, not run on hardware
//
#include <hip/hip_runtime.h>

// ---------------------------------------------------------------------------
// CDNA5 / gfx1250 wave32 WMMA types
// ---------------------------------------------------------------------------
typedef __attribute__((ext_vector_type(16))) __bf16 v16bf;
typedef __attribute__((ext_vector_type(8)))  float  v8f;

#define KD      48
#define GR      8
#define DKD     8
#define DVD     16
#define CINCH   64
#define EPSV    1e-5f
#define S_TOT   110592      // 48*48*48 spatial (h*2304 + w*48 + d)
#define BATCH   2304        // 48*48  (b = w*48 + d)
#define NPROJ   256         // 192 kv rows + 64 q rows

// workspace layout (float offsets); total ~56.7M floats (~227 MB)
#define OFF_PN        0
#define OFF_OUT       (256 * 110592)
#define OFF_EMB       (2 * 256 * 110592)
#define OFF_GRAM      (OFF_EMB + 32 * 2304)
#define OFF_XSUM      (OFF_GRAM + 64 * 64)
#define OFF_SIMSTATS  (OFF_XSUM + 64)
#define OFF_OUTSTATS  (OFF_SIMSTATS + 48)
#define OFF_PROJCOEF  (OFF_OUTSTATS + 512)
#define OFF_SIMCOEF   (OFF_PROJCOEF + 512)
#define OFF_OUTCOEF   (OFF_SIMCOEF + 32)
#define STATS_ZERO_FLOATS (64 * 64 + 64 + 48 + 512)   // gram..outstats contiguous

__device__ __forceinline__ float wave_sum32(float v) {
    for (int off = 16; off > 0; off >>= 1) v += __shfl_down(v, off, 32);
    return v;
}

// CDNA5 async global->LDS copy (ASYNCcnt tracked, no VGPR staging).
// lds_off must be the LDS byte offset (= low 32 bits of the flat shared addr).
__device__ __forceinline__ void async_load_b32(unsigned lds_off, const void* gaddr) {
    asm volatile("global_load_async_to_lds_b32 %0, %1, off"
                 :: "v"(lds_off), "v"((unsigned long long)(uintptr_t)gaddr)
                 : "memory");
}
__device__ __forceinline__ void async_wait0() {
    asm volatile("s_wait_asynccnt 0x0" ::: "memory");
}

// ---------------------------------------------------------------------------
// Kernel A: per-channel sum of x and 64x64 Gram  (for analytic proj-BN stats)
// ---------------------------------------------------------------------------
__global__ void __launch_bounds__(256) k_xstats(const float* __restrict__ x,
                                                float* __restrict__ gram,
                                                float* __restrict__ xsum) {
    __shared__ float xs[64][65];
    const int t  = threadIdx.x;
    const int s0 = blockIdx.x * 1024;
    const int ci = (t & 15) * 4;
    const int cj = (t >> 4) * 4;
    float acc[4][4] = {};
    float csum = 0.f;
    for (int st = 0; st < 16; ++st) {
        const int sb = s0 + st * 64;
        for (int l = t; l < 4096; l += 256)
            xs[l >> 6][l & 63] = x[(size_t)(l >> 6) * S_TOT + sb + (l & 63)];
        __syncthreads();
        for (int sl = 0; sl < 64; ++sl) {
            float a0 = xs[ci + 0][sl], a1 = xs[ci + 1][sl];
            float a2 = xs[ci + 2][sl], a3 = xs[ci + 3][sl];
            float b0 = xs[cj + 0][sl], b1 = xs[cj + 1][sl];
            float b2 = xs[cj + 2][sl], b3 = xs[cj + 3][sl];
            acc[0][0] += a0 * b0; acc[0][1] += a0 * b1; acc[0][2] += a0 * b2; acc[0][3] += a0 * b3;
            acc[1][0] += a1 * b0; acc[1][1] += a1 * b1; acc[1][2] += a1 * b2; acc[1][3] += a1 * b3;
            acc[2][0] += a2 * b0; acc[2][1] += a2 * b1; acc[2][2] += a2 * b2; acc[2][3] += a2 * b3;
            acc[3][0] += a3 * b0; acc[3][1] += a3 * b1; acc[3][2] += a3 * b2; acc[3][3] += a3 * b3;
        }
        if (t < 64) {
            float cs = 0.f;
            for (int sl = 0; sl < 64; ++sl) cs += xs[t][sl];
            csum += cs;
        }
        __syncthreads();
    }
    for (int a = 0; a < 4; ++a)
        for (int b = 0; b < 4; ++b)
            atomicAdd(&gram[(ci + a) * 64 + cj + b], acc[a][b]);
    if (t < 64) atomicAdd(&xsum[t], csum);
}

// ---------------------------------------------------------------------------
// Kernel B: fold projection BN into per-row scale/shift
//   mean_o = W[o].mu ;  var_o = W[o] Gram W[o]^T / S - mean^2
// ---------------------------------------------------------------------------
__global__ void k_projcoef(const float* __restrict__ W_kv, const float* __restrict__ W_q,
                           const float* __restrict__ g_kv, const float* __restrict__ b_kv,
                           const float* __restrict__ g_q,  const float* __restrict__ b_q,
                           const float* __restrict__ gram, const float* __restrict__ xsum,
                           float* __restrict__ projcoef) {
    __shared__ float mu[64];
    const int t = threadIdx.x;
    if (t < 64) mu[t] = xsum[t] / (float)S_TOT;
    __syncthreads();
    const float* Wr;
    float gg, bb;
    if (t < 192) { Wr = W_kv + t * 64;        gg = g_kv[t];       bb = b_kv[t]; }
    else         { Wr = W_q + (t - 192) * 64; gg = g_q[t - 192];  bb = b_q[t - 192]; }
    float m = 0.f;
    for (int c = 0; c < 64; ++c) m += Wr[c] * mu[c];
    float q = 0.f;
    for (int c = 0; c < 64; ++c) {
        float r = 0.f;
        const float* grow = gram + c * 64;
        for (int c2 = 0; c2 < 64; ++c2) r += Wr[c2] * grow[c2];
        q += Wr[c] * r;
    }
    float var = q / (float)S_TOT - m * m;
    float a = gg * rsqrtf(var + EPSV);
    projcoef[2 * t]     = a;
    projcoef[2 * t + 1] = bb - m * a;
}

// ---------------------------------------------------------------------------
// Kernel: gather relative embeddings  emb[ce][i*48+j] = relative[ce][fidx[i*48+j]]
// ---------------------------------------------------------------------------
__global__ void k_emb(const float* __restrict__ relative, const int* __restrict__ fidx,
                      float* __restrict__ emb) {
    int i = blockIdx.x * blockDim.x + threadIdx.x;
    if (i < 32 * 2304) {
        int ce = i / 2304, ij = i % 2304;
        emb[i] = relative[ce * 95 + fidx[ij]];
    }
}

// ---------------------------------------------------------------------------
// Kernel C: projection GEMM  PN(256 x S) = BN( W_all(256x64) @ X(64xS) )
// One wave owns a 16-column strip of X and loops over all 16 M-tiles:
// B (x columns) is loaded into registers ONCE, A (64KB weights, L1-hot) is
// re-fetched per tile -> 32 back-to-back bf16 WMMAs per wave, x read exactly
// once from HBM instead of 16x through L2.
// ---------------------------------------------------------------------------
__global__ void __launch_bounds__(256) k_proj_gemm(const float* __restrict__ x,
                                                   const float* __restrict__ W_kv,
                                                   const float* __restrict__ W_q,
                                                   const float* __restrict__ projcoef,
                                                   float* __restrict__ PN) {
    const int lane = threadIdx.x & 31;
    const int wid  = threadIdx.x >> 5;
    const int nt   = blockIdx.x * 8 + wid;          // 6912 n-strips
    const int half = lane >> 4;
    const int n    = lane & 15;                     // A row / B col / C col
    const int s0   = nt * 16;

    // B operands for K = 0..31 and 32..63 (loaded once per wave)
    v16bf b0, b1;
    for (int e = 0; e < 16; ++e) {
        int k = e + 8 * half + ((e >= 8) ? 8 : 0);
        b0[e] = (__bf16)x[(size_t)k * S_TOT + s0 + n];
        b1[e] = (__bf16)x[(size_t)(k + 32) * S_TOT + s0 + n];
    }

    for (int mt = 0; mt < 16; ++mt) {
        const int mrowA = mt * 16 + n;
        const float* Wr = (mrowA < 192) ? (W_kv + mrowA * 64) : (W_q + (mrowA - 192) * 64);
        v16bf a;
        v8f acc = {};
        for (int e = 0; e < 16; ++e) {              // A 16x32: lane row, K-pairs/VGPR
            int k = e + 8 * half + ((e >= 8) ? 8 : 0);
            a[e] = (__bf16)Wr[k];
        }
        acc = __builtin_amdgcn_wmma_f32_16x16x32_bf16(false, a, false, b0,
                                                      (short)0, acc, false, false);
        for (int e = 0; e < 16; ++e) {
            int k = 32 + e + 8 * half + ((e >= 8) ? 8 : 0);
            a[e] = (__bf16)Wr[k];
        }
        acc = __builtin_amdgcn_wmma_f32_16x16x32_bf16(false, a, false, b1,
                                                      (short)0, acc, false, false);
        for (int r = 0; r < 8; ++r) {               // C: M = r + 8*half
            int mrow = mt * 16 + r + 8 * half;
            float v = acc[r] * projcoef[2 * mrow] + projcoef[2 * mrow + 1];
            PN[(size_t)mrow * S_TOT + s0 + n] = v;
        }
    }
}

// ---------------------------------------------------------------------------
// Kernel D: per-(b,g) axial attention. mode 0: accumulate sim-BN stats.
// mode 1: logits -> softmax -> sv (WMMA) + sve (VALU) -> OUT + out-BN stats.
// One wave per (b,g); PN / emb are L2-resident. Tile loads use the CDNA5
// async global->LDS path (ASYNCcnt) instead of VGPR round-trips.
// ---------------------------------------------------------------------------
__global__ void __launch_bounds__(32) k_attn(const float* __restrict__ PN,
                                             const float* __restrict__ emb,
                                             const float* __restrict__ simcoef,
                                             float* __restrict__ simstats,
                                             float* __restrict__ OUT,
                                             float* __restrict__ outstats,
                                             const float* f_qr, const float* f_kr,
                                             const float* f_sv, const float* f_sve,
                                             int mode) {
    __shared__ float qs[8][48];
    __shared__ float ks[8][48];
    __shared__ float vs[16][48];
    __shared__ float sm[48][49];
    __shared__ float ob[32][48];

    const int bg   = blockIdx.x;
    const int b    = bg >> 3;
    const int g    = bg & 7;
    const int lane = threadIdx.x;
    const int half = lane >> 4;
    const int n    = lane & 15;
    const float fqr = *f_qr, fkr = *f_kr;

    // LDS byte offsets = low 32 bits of flat shared address (aperture trunc.)
    const unsigned qs_off = (unsigned)(uintptr_t)&qs[0][0];
    const unsigned ks_off = (unsigned)(uintptr_t)&ks[0][0];
    const unsigned vs_off = (unsigned)(uintptr_t)&vs[0][0];

    for (int idx = lane; idx < 8 * 48; idx += 32) {
        int c = idx / 48, i = idx % 48;
        async_load_b32(qs_off + 4u * idx,
                       PN + (size_t)(192 + g * 8 + c) * S_TOT + i * 2304 + b);
        async_load_b32(ks_off + 4u * idx,
                       PN + (size_t)(g * 24 + c) * S_TOT + i * 2304 + b);
    }
    for (int idx = lane; idx < 16 * 48; idx += 32) {
        int c = idx / 48, i = idx % 48;
        async_load_b32(vs_off + 4u * idx,
                       PN + (size_t)(g * 24 + 8 + c) * S_TOT + i * 2304 + b);
    }
    async_wait0();
    __syncthreads();

    float s1[3] = {0.f, 0.f, 0.f}, s2[3] = {0.f, 0.f, 0.f};
    float a_qk = 0.f, a_qr = 0.f, a_kr = 0.f, tsum = 0.f;
    if (mode == 1) {
        a_qk = simcoef[g * 4 + 0]; a_qr = simcoef[g * 4 + 1];
        a_kr = simcoef[g * 4 + 2]; tsum = simcoef[g * 4 + 3];
    }

    // qk = q^T k (K=8, zero-padded into K=32 bf16 WMMA), plus qr/kr relative terms
    for (int it = 0; it < 3; ++it) {
        for (int jt = 0; jt < 3; ++jt) {
            v16bf av, bv;
            for (int e = 0; e < 16; ++e) {
                int k = e + 8 * half + ((e >= 8) ? 8 : 0);
                av[e] = (k < 8) ? (__bf16)qs[k][it * 16 + n] : (__bf16)0.0f;
            }
            for (int e = 0; e < 16; ++e) {
                int k = e + 8 * half + ((e >= 8) ? 8 : 0);
                bv[e] = (k < 8) ? (__bf16)ks[k][jt * 16 + n] : (__bf16)0.0f;
            }
            v8f acc = {};
            acc = __builtin_amdgcn_wmma_f32_16x16x32_bf16(false, av, false, bv,
                                                          (short)0, acc, false, false);
            for (int r = 0; r < 8; ++r) {
                int i = it * 16 + r + 8 * half;
                int j = jt * 16 + n;
                float qk = acc[r];
                float qr = 0.f, kr = 0.f;
                const float* eq = emb + i * 48 + j;                // q_emb[c][i][j]
                const float* ek = emb + 8 * 2304 + j * 48 + i;     // k_emb[c][j][i]
                for (int c = 0; c < 8; ++c) {
                    qr += qs[c][i] * eq[c * 2304];
                    kr += ks[c][j] * ek[c * 2304];
                }
                qr *= fqr; kr *= fkr;
                if (mode == 0) {
                    s1[0] += qk; s2[0] += qk * qk;
                    s1[1] += qr; s2[1] += qr * qr;
                    s1[2] += kr; s2[2] += kr * kr;
                } else {
                    sm[i][j] = qk * a_qk + qr * a_qr + kr * a_kr + tsum;
                }
            }
        }
    }

    if (mode == 0) {
        for (int p = 0; p < 3; ++p) {
            float v1 = wave_sum32(s1[p]);
            float v2 = wave_sum32(s2[p]);
            if (lane == 0) {
                atomicAdd(&simstats[g * 6 + 2 * p],     v1);
                atomicAdd(&simstats[g * 6 + 2 * p + 1], v2);
            }
        }
        return;
    }

    __syncthreads();
    for (int i = lane; i < 48; i += 32) {                 // row softmax over j
        float mx = -1e30f;
        for (int j = 0; j < 48; ++j) mx = fmaxf(mx, sm[i][j]);
        float sum = 0.f;
        for (int j = 0; j < 48; ++j) { float e = __expf(sm[i][j] - mx); sm[i][j] = e; sum += e; }
        float inv = 1.0f / sum;
        for (int j = 0; j < 48; ++j) sm[i][j] *= inv;
    }
    __syncthreads();

    const float fsv = *f_sv, fsve = *f_sve;
    for (int it = 0; it < 3; ++it) {                      // sv: sim(48x48) @ v^T(48x16)
        v8f acc = {};
        for (int kc = 0; kc < 2; ++kc) {
            v16bf av, bv;
            for (int e = 0; e < 16; ++e) {
                int j = kc * 32 + e + 8 * half + ((e >= 8) ? 8 : 0);
                av[e] = (j < 48) ? (__bf16)sm[it * 16 + n][j] : (__bf16)0.0f;
            }
            for (int e = 0; e < 16; ++e) {
                int j = kc * 32 + e + 8 * half + ((e >= 8) ? 8 : 0);
                bv[e] = (j < 48) ? (__bf16)vs[n][j] : (__bf16)0.0f;
            }
            acc = __builtin_amdgcn_wmma_f32_16x16x32_bf16(false, av, false, bv,
                                                          (short)0, acc, false, false);
        }
        for (int r = 0; r < 8; ++r) {
            int i = it * 16 + r + 8 * half;
            int c = n;
            float sve = 0.f;
            const float* ev = emb + (16 + c) * 2304 + i * 48;
            for (int j = 0; j < 48; ++j) sve += sm[i][j] * ev[j];
            ob[c * 2 + 0][i] = acc[r] * fsv;
            ob[c * 2 + 1][i] = sve * fsve;
        }
    }
    __syncthreads();

    for (int cl = 0; cl < 32; ++cl) {
        int ch = g * 32 + cl;                              // (g*DV+c)*2 + s
        float p1 = 0.f, p2 = 0.f;
        for (int i = lane; i < 48; i += 32) { float v = ob[cl][i]; p1 += v; p2 += v * v; }
        p1 = wave_sum32(p1); p2 = wave_sum32(p2);
        if (lane == 0) {
            atomicAdd(&outstats[2 * ch],     p1);
            atomicAdd(&outstats[2 * ch + 1], p2);
        }
        for (int i = lane; i < 48; i += 32)
            OUT[(size_t)ch * S_TOT + b * 48 + i] = ob[cl][i];
    }
}

// ---------------------------------------------------------------------------
// Kernel E: sim-BN coefficients (per g: a_qk, a_qr, a_kr, sum of shifts)
// ---------------------------------------------------------------------------
__global__ void k_simcoef(const float* __restrict__ simstats,
                          const float* __restrict__ g_sim, const float* __restrict__ b_sim,
                          float* __restrict__ simcoef) {
    int g = threadIdx.x;
    if (g >= 8) return;
    const float cnt = (float)BATCH * 48.0f * 48.0f;
    float tacc = 0.f;
    for (int p = 0; p < 3; ++p) {
        float s  = simstats[g * 6 + 2 * p];
        float ss = simstats[g * 6 + 2 * p + 1];
        float m  = s / cnt;
        float var = ss / cnt - m * m;
        int ch = p * 8 + g;
        float a = g_sim[ch] * rsqrtf(var + EPSV);
        simcoef[g * 4 + p] = a;
        tacc += b_sim[ch] - m * a;
    }
    simcoef[g * 4 + 3] = tacc;
}

// ---------------------------------------------------------------------------
// Kernel F: out-BN coefficients (256 channels)
// ---------------------------------------------------------------------------
__global__ void k_outcoef(const float* __restrict__ outstats,
                          const float* __restrict__ g_out, const float* __restrict__ b_out,
                          float* __restrict__ outcoef) {
    int ch = threadIdx.x;
    const float cnt = (float)S_TOT;
    float s = outstats[2 * ch], ss = outstats[2 * ch + 1];
    float m = s / cnt;
    float var = ss / cnt - m * m;
    float a = g_out[ch] * rsqrtf(var + EPSV);
    outcoef[2 * ch]     = a;
    outcoef[2 * ch + 1] = b_out[ch] - m * a;
}

// ---------------------------------------------------------------------------
// Kernel G: normalize, sum sv+sve channels, LDS-transpose to (gc, h, w, d)
// ---------------------------------------------------------------------------
__global__ void __launch_bounds__(256) k_final(const float* __restrict__ OUT,
                                               const float* __restrict__ outcoef,
                                               float* __restrict__ out) {
    __shared__ float tile[48][49];
    const int gc = blockIdx.x / 48;
    const int w  = blockIdx.x % 48;
    const int t  = threadIdx.x;
    for (int s = 0; s < 2; ++s) {
        int ch = gc * 2 + s;
        float a = outcoef[2 * ch], tt = outcoef[2 * ch + 1];
        for (int e = t; e < 2304; e += 256) {
            int d = e / 48, i = e % 48;                     // contiguous reads over i
            float v = OUT[(size_t)ch * S_TOT + (w * 48 + d) * 48 + i] * a + tt;
            if (s == 0) tile[d][i] = v; else tile[d][i] += v;
        }
        __syncthreads();
    }
    for (int e = t; e < 2304; e += 256) {
        int i = e / 48, d = e % 48;                         // contiguous writes over d
        out[(size_t)gc * S_TOT + i * 2304 + w * 48 + d] = tile[d][i];
    }
}

// ---------------------------------------------------------------------------
extern "C" void kernel_launch(void* const* d_in, const int* in_sizes, int n_in,
                              void* d_out, int out_size, void* d_ws, size_t ws_size,
                              hipStream_t stream) {
    const float* x        = (const float*)d_in[0];
    const float* W_kv     = (const float*)d_in[1];
    const float* W_q      = (const float*)d_in[2];
    const float* g_kv     = (const float*)d_in[3];
    const float* b_kv     = (const float*)d_in[4];
    const float* g_q      = (const float*)d_in[5];
    const float* b_q      = (const float*)d_in[6];
    const float* g_sim    = (const float*)d_in[7];
    const float* b_sim    = (const float*)d_in[8];
    const float* g_out    = (const float*)d_in[9];
    const float* b_out    = (const float*)d_in[10];
    const float* relative = (const float*)d_in[11];
    const float* f_qr     = (const float*)d_in[12];
    const float* f_kr     = (const float*)d_in[13];
    const float* f_sv     = (const float*)d_in[14];
    const float* f_sve    = (const float*)d_in[15];
    const int*   fidx     = (const int*)d_in[16];

    float* ws       = (float*)d_ws;
    float* PN       = ws + OFF_PN;
    float* OUT      = ws + OFF_OUT;
    float* emb      = ws + OFF_EMB;
    float* gram     = ws + OFF_GRAM;
    float* xsum     = ws + OFF_XSUM;
    float* simstats = ws + OFF_SIMSTATS;
    float* outstats = ws + OFF_OUTSTATS;
    float* projcoef = ws + OFF_PROJCOEF;
    float* simcoef  = ws + OFF_SIMCOEF;
    float* outcoef  = ws + OFF_OUTCOEF;

    hipMemsetAsync(gram, 0, STATS_ZERO_FLOATS * sizeof(float), stream);

    k_xstats  <<<S_TOT / 1024, 256, 0, stream>>>(x, gram, xsum);
    k_projcoef<<<1, 256, 0, stream>>>(W_kv, W_q, g_kv, b_kv, g_q, b_q, gram, xsum, projcoef);
    k_emb     <<<(32 * 2304 + 255) / 256, 256, 0, stream>>>(relative, fidx, emb);
    k_proj_gemm<<<6912 / 8, 256, 0, stream>>>(x, W_kv, W_q, projcoef, PN);
    k_attn    <<<BATCH * GR, 32, 0, stream>>>(PN, emb, simcoef, simstats, OUT, outstats,
                                              f_qr, f_kr, f_sv, f_sve, 0);
    k_simcoef <<<1, 32, 0, stream>>>(simstats, g_sim, b_sim, simcoef);
    k_attn    <<<BATCH * GR, 32, 0, stream>>>(PN, emb, simcoef, simstats, OUT, outstats,
                                              f_qr, f_kr, f_sv, f_sve, 1);
    k_outcoef <<<1, 256, 0, stream>>>(outstats, g_out, b_out, outcoef);
    k_final   <<<128 * 48, 256, 0, stream>>>(OUT, outcoef, (float*)d_out);
}
